// BidirectionAttention_33389075759302
// MI455X (gfx1250) — matrix operand order repaired and checked by reading, not compiled, not run on hardware
//
#include <hip/hip_runtime.h>
#include <hip/hip_bf16.h>

// Problem constants (from reference)
#define Bc      2
#define Cch     64
#define Nn      32768          // D*H*W feat tokens
#define MAPD    128
#define Mm      512            // map tokens
#define NHEADS  4
#define DHd     64
#define INNERd  256
#define OUTD    64
#define BH      (Bc*NHEADS)    // 8
#define NG      32             // column-pass row groups (deterministic partials)
#define SCALEF  0.125f

typedef __attribute__((ext_vector_type(8)))  __bf16 bf16x8;
typedef __attribute__((ext_vector_type(16))) __bf16 bf16x16;
typedef __attribute__((ext_vector_type(8)))  float  f32x8;

static __device__ __forceinline__ f32x8 wmma_bf16(bf16x16 a, bf16x16 b, f32x8 c) {
  // D = A(16x32 bf16) * B(32x16 bf16) + C(16x16 f32)
  return __builtin_amdgcn_wmma_f32_16x16x32_bf16(false, a, false, b, (short)0, c, false, false);
}

static __device__ __forceinline__ bf16x16 cat8(bf16x8 lo, bf16x8 hi) {
  bf16x16 r;
#pragma unroll
  for (int i = 0; i < 8; ++i) { r[i] = lo[i]; r[i + 8] = hi[i]; }
  return r;
}

// A-operand (16-bit 16x32): lane holds row M=lane%16; elements e -> K = kb+16*lh+e
// (contiguous, matching B)  (lh = lane/16)
static __device__ __forceinline__ bf16x16 load_a16(const __bf16* row, int kb, int lh) {
  bf16x8 lo = *(const bf16x8*)(row + kb + 16 * lh);
  bf16x8 hi = *(const bf16x8*)(row + kb + 16 * lh + 8);
  return cat8(lo, hi);
}
static __device__ __forceinline__ bf16x16 load_a16_nt(const __bf16* row, int kb, int lh) {
  bf16x8 lo = __builtin_nontemporal_load((const bf16x8*)(row + kb + 16 * lh));
  bf16x8 hi = __builtin_nontemporal_load((const bf16x8*)(row + kb + 16 * lh + 8));
  return cat8(lo, hi);
}
// B-operand (16-bit 32x16): lane holds col N=lane%16; elements e -> K = 16*lh + e (contiguous)
static __device__ __forceinline__ bf16x16 load_b16(const __bf16* p) {
  bf16x8 lo = *(const bf16x8*)(p);
  bf16x8 hi = *(const bf16x8*)(p + 8);
  return cat8(lo, hi);
}
static __device__ __forceinline__ bf16x16 load_b16_nt(const __bf16* p) {
  bf16x8 lo = __builtin_nontemporal_load((const bf16x8*)p);
  bf16x8 hi = __builtin_nontemporal_load((const bf16x8*)p + 1);
  return cat8(lo, hi);
}

static __device__ __forceinline__ void atomicMaxF(float* addr, float val) {
  int cur = __float_as_int(*(volatile float*)addr);
  while (__int_as_float(cur) < val) {
    int assumed = cur;
    cur = atomicCAS((int*)addr, assumed, __float_as_int(val));
    if (cur == assumed) break;
  }
}

// ---------------------------------------------------------------- init colmax
__global__ void k_init_colmax(float* __restrict__ colmax) {
  int i = blockIdx.x * 256 + threadIdx.x;
  if (i < BH * Mm) colmax[i] = -3.0e38f;
}

// ------------------------------------------------- feat projection (1x1 conv)
// Qf[bh][n][d] = SCALE * sum_c w[d*4+h][c] * feat[b][c][n]   (bf16)
// VfT[bh][d][n] = sum_c w[256 + d*4+h][c] * feat[b][c][n]    (bf16, transposed)
__global__ __launch_bounds__(256)
void k_proj_feat(const float* __restrict__ feat, const float* __restrict__ w,
                 __bf16* __restrict__ Qf, __bf16* __restrict__ VfT) {
  int b = blockIdx.y;
  int n = blockIdx.x * 256 + threadIdx.x;
  float fv[Cch];
#pragma unroll
  for (int c = 0; c < Cch; ++c) fv[c] = feat[((size_t)(b * Cch + c)) * Nn + n];
  for (int h = 0; h < NHEADS; ++h) {
    int bh = b * NHEADS + h;
    for (int dblk = 0; dblk < DHd / 8; ++dblk) {
      bf16x8 pq;
      for (int dd = 0; dd < 8; ++dd) {
        int d = dblk * 8 + dd;
        int o = d * NHEADS + h;
        float q = 0.f, v = 0.f;
#pragma unroll
        for (int c = 0; c < Cch; ++c) {
          q += w[(size_t)o * Cch + c] * fv[c];
          v += w[(size_t)(o + INNERd) * Cch + c] * fv[c];
        }
        pq[dd] = (__bf16)(q * SCALEF);
        VfT[((size_t)(bh * DHd + d)) * Nn + n] = (__bf16)v;
      }
      *(bf16x8*)(Qf + ((size_t)bh * Nn + n) * DHd + dblk * 8) = pq;
    }
  }
}

// -------------------------------------------------- map projection (1x1 conv)
__global__ __launch_bounds__(128)
void k_proj_map(const float* __restrict__ smap, const float* __restrict__ w,
                __bf16* __restrict__ Qm, __bf16* __restrict__ VmT) {
  int b = blockIdx.x / Mm;
  int j = blockIdx.x % Mm;
  __shared__ float mv[MAPD];
  int tid = threadIdx.x;
  mv[tid] = smap[((size_t)(b * MAPD + tid)) * Mm + j];
  __syncthreads();
#pragma unroll
  for (int r = 0; r < 4; ++r) {
    int o = tid + r * 128;
    float s = 0.f;
#pragma unroll
    for (int c = 0; c < MAPD; ++c) s += w[(size_t)o * MAPD + c] * mv[c];
    if (o < INNERd) {
      int d = o >> 2, h = o & 3;
      Qm[((size_t)((b * NHEADS + h) * Mm + j)) * DHd + d] = (__bf16)s;
    } else {
      int o2 = o - INNERd;
      int d = o2 >> 2, h = o2 & 3;
      VmT[((size_t)((b * NHEADS + h) * DHd + d)) * Mm + j] = (__bf16)s;
    }
  }
}

// ------------------------- pass 1: row softmax + feat_out tile + global colmax
// grid (Nn/64, BH), 128 threads (4 waves). Each wave owns 16 rows.
__global__ __launch_bounds__(128)
void k_attn_row(const __bf16* __restrict__ Qf, const __bf16* __restrict__ Qm,
                const __bf16* __restrict__ VmT, float* __restrict__ colmax,
                __bf16* __restrict__ merged) {
  constexpr int SLD = 520;                    // padded row stride (bf16 elems)
  __shared__ __bf16 Sl[64 * SLD];
  __shared__ float rowmax_l[64];
  __shared__ float psum[128];
  int bh = blockIdx.y, tile = blockIdx.x, tid = threadIdx.x;
  int wave = tid >> 5, lane = tid & 31, lh = lane >> 4, ln = lane & 15;

  // ---- S = Qf * Qm^T (per-head), track running row max + per-tile col max
  const __bf16* qrow = Qf + ((size_t)bh * Nn + (size_t)tile * 64 + wave * 16 + ln) * DHd;
  bf16x16 a0 = load_a16(qrow, 0, lh);
  bf16x16 a1 = load_a16(qrow, 32, lh);
  const __bf16* kbase = Qm + (size_t)bh * Mm * DHd + 16 * lh;
  float rm[8];
#pragma unroll
  for (int g = 0; g < 8; ++g) rm[g] = -3.0e38f;
  // software-pipelined B operands
  bf16x16 b0 = load_b16(kbase + (size_t)ln * DHd);
  bf16x16 b1 = load_b16(kbase + (size_t)ln * DHd + 32);
  for (int ct = 0; ct < Mm / 16; ++ct) {
    int j0 = ct * 16;
    int nj = (ct < Mm / 16 - 1) ? (j0 + 16) : j0;   // clamped prefetch index
    const __bf16* nk = kbase + (size_t)(nj + ln) * DHd;
    bf16x16 nb0 = load_b16(nk);
    bf16x16 nb1 = load_b16(nk + 32);
    f32x8 s = {0.f, 0.f, 0.f, 0.f, 0.f, 0.f, 0.f, 0.f};
    s = wmma_bf16(a0, b0, s);
    s = wmma_bf16(a1, b1, s);
    float cm = -3.0e38f;
#pragma unroll
    for (int g = 0; g < 8; ++g) {
      float v = s[g];
      rm[g] = fmaxf(rm[g], v);
      cm = fmaxf(cm, v);
      int il = wave * 16 + g + 8 * lh;       // C/D layout: M = g + 8*(lane/16)
      Sl[il * SLD + j0 + ln] = (__bf16)v;    // N = lane%16
    }
    cm = fmaxf(cm, __shfl_xor(cm, 16, 32));
    if (lane < 16) atomicMaxF(&colmax[bh * Mm + j0 + lane], cm);
    b0 = nb0; b1 = nb1;
  }
#pragma unroll
  for (int g = 0; g < 8; ++g) {
    float v = rm[g];
    v = fmaxf(v, __shfl_xor(v, 1, 32));
    v = fmaxf(v, __shfl_xor(v, 2, 32));
    v = fmaxf(v, __shfl_xor(v, 4, 32));
    v = fmaxf(v, __shfl_xor(v, 8, 32));
    if (ln == 0) rowmax_l[wave * 16 + g + 8 * lh] = v;
  }
  __syncthreads();

  // ---- row softmax in LDS (2 threads per row)
  int il = tid & 63, half = tid >> 6;
  float rmv = rowmax_l[il];
  float ssum = 0.f;
  for (int jj = half * 256; jj < half * 256 + 256; ++jj)
    ssum += __expf((float)Sl[il * SLD + jj] - rmv);
  psum[half * 64 + il] = ssum;
  __syncthreads();
  float inv = 1.f / (psum[il] + psum[64 + il]);
  for (int jj = half * 256; jj < half * 256 + 256; ++jj)
    Sl[il * SLD + jj] = (__bf16)(__expf((float)Sl[il * SLD + jj] - rmv) * inv);
  __syncthreads();

  // ---- feat_out = P(64x512) * Vm(512x64), write merged-head bf16 (pipelined)
  f32x8 acc[4];
#pragma unroll
  for (int dt = 0; dt < 4; ++dt) acc[dt] = (f32x8){0.f, 0.f, 0.f, 0.f, 0.f, 0.f, 0.f, 0.f};
  const __bf16* prow = Sl + (wave * 16 + ln) * SLD;
  const __bf16* vbase = VmT + (size_t)bh * DHd * Mm + 16 * lh;
  bf16x16 ap = load_a16(prow, 0, lh);
  bf16x16 bv[4];
#pragma unroll
  for (int dt = 0; dt < 4; ++dt)
    bv[dt] = load_b16(vbase + (size_t)(dt * 16 + ln) * Mm);
  for (int kb = 0; kb < Mm; kb += 32) {
    int nkb = (kb < Mm - 32) ? (kb + 32) : kb;      // clamped prefetch
    bf16x16 nap = load_a16(prow, nkb, lh);
    bf16x16 nbv[4];
#pragma unroll
    for (int dt = 0; dt < 4; ++dt)
      nbv[dt] = load_b16(vbase + (size_t)(dt * 16 + ln) * Mm + nkb);
#pragma unroll
    for (int dt = 0; dt < 4; ++dt)
      acc[dt] = wmma_bf16(ap, bv[dt], acc[dt]);
    ap = nap;
#pragma unroll
    for (int dt = 0; dt < 4; ++dt) bv[dt] = nbv[dt];
  }
  int b = bh >> 2, h = bh & 3;
#pragma unroll
  for (int dt = 0; dt < 4; ++dt)
#pragma unroll
    for (int g = 0; g < 8; ++g) {
      int iloc = wave * 16 + g + 8 * lh;
      int n = tile * 64 + iloc;
      int d = dt * 16 + ln;
      merged[((size_t)b * INNERd + d * NHEADS + h) * Nn + n] = (__bf16)acc[dt][g];
    }
}

// ------- pass 2: column softmax numerators + map accumulation (deterministic)
// grid (NG, BH), 256 threads (8 waves). wave = (row-quarter, col-half).
__global__ __launch_bounds__(256)
void k_attn_col(const __bf16* __restrict__ Qf, const __bf16* __restrict__ Qm,
                const __bf16* __restrict__ VfT, const float* __restrict__ colmax,
                float* __restrict__ colsum_part, float* __restrict__ Macc_part) {
  constexpr int ETD = 72;                    // padded col stride of E^T (bf16)
  __shared__ __bf16 Et[Mm * ETD];            // E^T: [j][i_local], 72 KB
  __shared__ float cspart[8 * Mm];           // per-wave colsum partials
  int bh = blockIdx.y, ng = blockIdx.x, tid = threadIdx.x;
  int wave = tid >> 5, lane = tid & 31, lh = lane >> 4, ln = lane & 15;
  for (int i = tid; i < 8 * Mm; i += 256) cspart[i] = 0.f;
  __syncthreads();

  f32x8 acc[4][4];
#pragma unroll
  for (int jt = 0; jt < 4; ++jt)
#pragma unroll
    for (int dt = 0; dt < 4; ++dt)
      acc[jt][dt] = (f32x8){0.f, 0.f, 0.f, 0.f, 0.f, 0.f, 0.f, 0.f};

  int rw16 = 16 * (wave & 3);
  int ch = wave >> 2;
  const __bf16* kbase = Qm + (size_t)bh * Mm * DHd + 16 * lh;
  for (int cc = 0; cc < (Nn / NG) / 64; ++cc) {      // chunks of 64 rows
    int Rbase = ng * (Nn / NG) + cc * 64;
    // phase A: E = exp(S - colmax) into E^T LDS; per-wave colsum partials
    const __bf16* qrow = Qf + ((size_t)bh * Nn + Rbase + rw16 + ln) * DHd;
    bf16x16 a0 = load_a16_nt(qrow, 0, lh);           // Qf: streamed, read-once here
    bf16x16 a1 = load_a16_nt(qrow, 32, lh);
    int ct0 = ch * 16;
    bf16x16 b0 = load_b16(kbase + (size_t)(ct0 * 16 + ln) * DHd);
    bf16x16 b1 = load_b16(kbase + (size_t)(ct0 * 16 + ln) * DHd + 32);
    for (int ct = ct0; ct < ct0 + 16; ++ct) {
      int j0 = ct * 16;
      int nj = (ct < ct0 + 15) ? (j0 + 16) : j0;     // clamped prefetch
      const __bf16* nk = kbase + (size_t)(nj + ln) * DHd;
      bf16x16 nb0 = load_b16(nk);
      bf16x16 nb1 = load_b16(nk + 32);
      f32x8 s = {0.f, 0.f, 0.f, 0.f, 0.f, 0.f, 0.f, 0.f};
      s = wmma_bf16(a0, b0, s);
      s = wmma_bf16(a1, b1, s);
      float cm = colmax[bh * Mm + j0 + ln];
      float cs = 0.f;
#pragma unroll
      for (int g = 0; g < 8; ++g) {
        float e = __expf(s[g] - cm);
        cs += e;
        int ilc = rw16 + g + 8 * lh;
        Et[(j0 + ln) * ETD + ilc] = (__bf16)e;
      }
      cs += __shfl_xor(cs, 16, 32);
      if (lane < 16) cspart[wave * Mm + j0 + lane] += cs;
      b0 = nb0; b1 = nb1;
    }
    __syncthreads();
    // phase B: Macc += E^T(512x64) * Vf(64x64); wave owns 64 j-rows
    bf16x16 bv[2][4];                                 // both K-steps up front
#pragma unroll
    for (int kk = 0; kk < 2; ++kk)
#pragma unroll
      for (int dt = 0; dt < 4; ++dt)
        bv[kk][dt] = load_b16_nt(VfT + ((size_t)bh * DHd + dt * 16 + ln) * Nn +
                                 Rbase + kk * 32 + 16 * lh);
#pragma unroll
    for (int kk = 0; kk < 2; ++kk) {
#pragma unroll
      for (int jt = 0; jt < 4; ++jt) {
        const __bf16* erow = Et + (wave * 64 + jt * 16 + ln) * ETD;
        bf16x16 ae = load_a16(erow, kk * 32, lh);
#pragma unroll
        for (int dt = 0; dt < 4; ++dt)
          acc[jt][dt] = wmma_bf16(ae, bv[kk][dt], acc[jt][dt]);
      }
    }
    __syncthreads();
  }
  // colsum partial for this row-group (fixed order -> deterministic)
  for (int j = tid; j < Mm; j += 256) {
    float s = 0.f;
#pragma unroll
    for (int w2 = 0; w2 < 8; ++w2) s += cspart[w2 * Mm + j];
    colsum_part[((size_t)bh * NG + ng) * Mm + j] = s;
  }
#pragma unroll
  for (int jt = 0; jt < 4; ++jt)
#pragma unroll
    for (int dt = 0; dt < 4; ++dt)
#pragma unroll
      for (int g = 0; g < 8; ++g) {
        int j = wave * 64 + jt * 16 + g + 8 * lh;
        int d = dt * 16 + ln;
        Macc_part[(((size_t)bh * NG + ng) * Mm + j) * DHd + d] = acc[jt][dt][g];
      }
}

// -------------------------------- reduce NG partials (fixed order, no atomics)
__global__ void k_reduce(const float* __restrict__ Macc_part,
                         const float* __restrict__ colsum_part,
                         float* __restrict__ Mn, float* __restrict__ colsum_tot) {
  int idx = blockIdx.x * 256 + threadIdx.x;          // BH*Mm*DHd = 262144
  int bhj = idx / DHd, d = idx % DHd;
  int bh = bhj / Mm, j = bhj % Mm;
  float s = 0.f;
#pragma unroll
  for (int ng = 0; ng < NG; ++ng)
    s += Macc_part[(((size_t)bh * NG + ng) * Mm + j) * DHd + d];
  Mn[idx] = s;
  if (idx < BH * Mm) {
    int bh2 = idx / Mm, j2 = idx % Mm;
    float cs = 0.f;
#pragma unroll
    for (int ng = 0; ng < NG; ++ng)
      cs += colsum_part[((size_t)bh2 * NG + ng) * Mm + j2];
    colsum_tot[idx] = cs;
  }
}

// ------------------------------------------- map output projection (to d_out)
__global__ __launch_bounds__(256)
void k_map_out(const float* __restrict__ Mn, const float* __restrict__ colsum_tot,
               const float* __restrict__ w, float* __restrict__ outp) {
  int b = blockIdx.x / MAPD;
  int om = blockIdx.x % MAPD;
  int tid = threadIdx.x;
  for (int jj = 0; jj < 2; ++jj) {
    int j = jj * 256 + tid;
    float inv[NHEADS];
#pragma unroll
    for (int h = 0; h < NHEADS; ++h)
      inv[h] = 1.f / colsum_tot[(b * NHEADS + h) * Mm + j];
    float s = 0.f;
#pragma unroll 4
    for (int o = 0; o < INNERd; ++o) {
      int d = o >> 2, h = o & 3;
      s += w[(size_t)om * INNERd + o] *
           Mn[((size_t)(b * NHEADS + h) * Mm + j) * DHd + d] * inv[h];
    }
    outp[((size_t)(b * MAPD + om)) * Mm + j] = s;
  }
}

// ------------------------------------------ feat output projection (to d_out)
__global__ __launch_bounds__(256)
void k_feat_out(const __bf16* __restrict__ merged, const float* __restrict__ w,
                float* __restrict__ outp) {
  int b = blockIdx.y / OUTD;
  int of = blockIdx.y % OUTD;
  int n = blockIdx.x * 256 + threadIdx.x;
  float s = 0.f;
#pragma unroll 8
  for (int o = 0; o < INNERd; ++o)
    s += w[(size_t)of * INNERd + o] * (float)merged[((size_t)b * INNERd + o) * Nn + n];
  outp[((size_t)(b * OUTD + of)) * Nn + n] = s;
}

extern "C" void kernel_launch(void* const* d_in, const int* in_sizes, int n_in,
                              void* d_out, int out_size, void* d_ws, size_t ws_size,
                              hipStream_t stream) {
  (void)in_sizes; (void)n_in; (void)out_size; (void)ws_size;
  const float* feat       = (const float*)d_in[0];
  const float* smap       = (const float*)d_in[1];
  const float* w_feat_qv  = (const float*)d_in[2];
  const float* w_map_qv   = (const float*)d_in[3];
  const float* w_feat_out = (const float*)d_in[4];
  const float* w_map_out  = (const float*)d_in[5];
  float* out = (float*)d_out;

  char* ws = (char*)d_ws;
  size_t off = 0;
  auto carve = [&](size_t bytes) -> void* {
    void* p = ws + off;
    off += (bytes + 255) & ~(size_t)255;
    return p;
  };
  __bf16* Qf          = (__bf16*)carve((size_t)BH * Nn * DHd * 2);   // 33.5 MB
  __bf16* VfT         = (__bf16*)carve((size_t)BH * DHd * Nn * 2);   // 33.5 MB
  __bf16* Qm          = (__bf16*)carve((size_t)BH * Mm * DHd * 2);
  __bf16* VmT         = (__bf16*)carve((size_t)BH * DHd * Mm * 2);
  __bf16* merged      = (__bf16*)carve((size_t)Bc * INNERd * Nn * 2); // 33.5 MB
  float*  colmax      = (float*)carve((size_t)BH * Mm * 4);
  float*  colsum_part = (float*)carve((size_t)BH * NG * Mm * 4);
  float*  Macc_part   = (float*)carve((size_t)BH * NG * Mm * DHd * 4); // 33.5 MB
  float*  Mn          = (float*)carve((size_t)BH * Mm * DHd * 4);
  float*  colsum_tot  = (float*)carve((size_t)BH * Mm * 4);

  k_init_colmax<<<(BH * Mm + 255) / 256, 256, 0, stream>>>(colmax);
  k_proj_feat<<<dim3(Nn / 256, Bc), 256, 0, stream>>>(feat, w_feat_qv, Qf, VfT);
  k_proj_map<<<Bc * Mm, 128, 0, stream>>>(smap, w_map_qv, Qm, VmT);
  k_attn_row<<<dim3(Nn / 64, BH), 128, 0, stream>>>(Qf, Qm, VmT, colmax, merged);
  k_attn_col<<<dim3(NG, BH), 256, 0, stream>>>(Qf, Qm, VfT, colmax, colsum_part, Macc_part);
  k_reduce<<<(BH * Mm * DHd + 255) / 256, 256, 0, stream>>>(Macc_part, colsum_part, Mn, colsum_tot);
  k_map_out<<<Bc * MAPD, 256, 0, stream>>>(Mn, colsum_tot, w_map_out,
                                           out + (size_t)Bc * OUTD * Nn);
  k_feat_out<<<dim3(Nn / 256, Bc * OUTD), 256, 0, stream>>>(merged, w_feat_out, out);
}
